// CohortAwareBlock_45535243272315
// MI455X (gfx1250) — matrix operand order repaired and checked by reading, not compiled, not run on hardware
//
#include <hip/hip_runtime.h>
#include <hip/hip_bf16.h>
#include <stdint.h>

typedef __bf16 bf16_t;
typedef __attribute__((ext_vector_type(8)))  __bf16 v8bf;
typedef __attribute__((ext_vector_type(16))) __bf16 v16bf;
typedef __attribute__((ext_vector_type(8)))  float  v8f;
typedef __attribute__((ext_vector_type(4)))  int    v4i;

#define DIMD   1024
#define NHEADS 16
#define HD     64
#define BATCH  8
#define SEQ    1024
#define EQKV   3072
#define QSCALE 0.125f   // 64^-0.5

#if defined(__has_builtin)
#  if __has_builtin(__builtin_amdgcn_global_load_async_to_lds_b128)
#    define HAVE_ASYNC_LDS 1
#  endif
#endif
#ifndef HAVE_ASYNC_LDS
#  define HAVE_ASYNC_LDS 0
#endif

static __device__ __forceinline__ v16bf frag16(const bf16_t* p0, const bf16_t* p1) {
  v8bf a = *(const v8bf*)p0;
  v8bf b = *(const v8bf*)p1;
  return __builtin_shufflevector(a, b, 0,1,2,3,4,5,6,7,8,9,10,11,12,13,14,15);
}

static __device__ __forceinline__ v8f wmma_bf16(v16bf a, v16bf b, v8f c) {
  // D = A(16x32 bf16) * B(32x16 bf16) + C(16x16 f32)
  return __builtin_amdgcn_wmma_f32_16x16x32_bf16(false, a, false, b, (short)0, c, false, false);
}

// 16-byte global -> LDS copy; async (ASYNCcnt-tracked) when available.
static __device__ __forceinline__ void cp16_g2l(const bf16_t* g, bf16_t* l) {
#if HAVE_ASYNC_LDS
  typedef __attribute__((address_space(1))) v4i gv4i;
  typedef __attribute__((address_space(3))) v4i lv4i;
  __builtin_amdgcn_global_load_async_to_lds_b128(
      (gv4i*)(uintptr_t)g, (lv4i*)(uintptr_t)l, 0, 0);
#else
  *(v8bf*)l = *(const v8bf*)g;
#endif
}

static __device__ __forceinline__ void wait_async0() {
#if HAVE_ASYNC_LDS
#  if defined(__has_builtin) && __has_builtin(__builtin_amdgcn_s_wait_asynccnt)
  __builtin_amdgcn_s_wait_asynccnt(0);
#  else
  __asm__ volatile("s_wait_asynccnt 0x0" ::: "memory");
#  endif
#endif
}

// ---------------------------------------------------------------- convert
__global__ void f32_to_bf16_kernel(const float* __restrict__ s,
                                   bf16_t* __restrict__ d, int n) {
  int i = blockIdx.x * blockDim.x + threadIdx.x;
  if (i < n) d[i] = (bf16_t)s[i];
}

// ---------------------------------------------------------------- fused QKV GEMM
// grid (EQKV/64=48, SEQ/256=4, B=8), block 256 (8 waves).
// Wave computes 32 rows x 64 cols (2 M-tiles x 4 N-tiles).
// B (weight) tile staged in LDS once per workgroup per K-step, double-buffered
// via async global->LDS DMA; row stride padded to 40 elems for bank spread.
__global__ __launch_bounds__(256) void qkv_gemm_kernel(
    const bf16_t* __restrict__ xb,                       // [B,N,DIM] bf16
    const bf16_t* __restrict__ kvw, const float* __restrict__ kvb,
    const bf16_t* __restrict__ sqw, const float* __restrict__ sqb,
    const bf16_t* __restrict__ cqw, const float* __restrict__ cqb,
    const int*    __restrict__ cid,
    bf16_t* __restrict__ qbuf,    // [B*H, N, 64] (pre-scaled)
    bf16_t* __restrict__ kbuf,    // [B*H, N, 64]
    bf16_t* __restrict__ vT)      // [B*H, 64, N]
{
  __shared__ __align__(16) bf16_t bstage[2][64][40];

  const int tid  = threadIdx.x;
  const int lane = tid & 31;
  const int wave = tid >> 5;
  const int ln   = lane & 15;
  const int hi   = lane >> 4;            // half-wave select
  const int b    = blockIdx.z;
  const int eb   = blockIdx.x * 64;
  const int row0 = blockIdx.y * 256 + wave * 32;

  // region: [0,2048) KV | [2048,3008) shared-Q | [3008,3072) cohort-Q
  const int region = (eb < 2048) ? 0 : (eb < 3008 ? 1 : 2);

  // per-thread staging assignment: row srow of the 64-row weight tile, 8-elem chunk
  const int srow   = tid >> 2;          // 0..63
  const int schunk = (tid & 3) * 8;     // 0,8,16,24
  const bf16_t* wsrc;
  {
    int es = eb + srow;
    if (region == 0)      wsrc = kvw + (size_t)es * DIMD;
    else if (region == 1) wsrc = sqw + (size_t)(es - 2048) * DIMD;
    else { int cc = cid[b]; wsrc = cqw + (size_t)(cc * 64 + es - 3008) * DIMD; }
  }

  float bias[4];
#pragma unroll
  for (int nt = 0; nt < 4; ++nt) {
    int e = eb + nt * 16 + ln;
    if (region == 0)      bias[nt] = kvb[e];
    else if (region == 1) bias[nt] = sqb[e - 2048];
    else { int cc = cid[b]; bias[nt] = cqb[cc * 64 + (e - 3008)]; }
  }

  v8f acc[2][4];
#pragma unroll
  for (int mt = 0; mt < 2; ++mt)
#pragma unroll
    for (int nt = 0; nt < 4; ++nt)
#pragma unroll
      for (int r = 0; r < 8; ++r) acc[mt][nt][r] = bias[nt];

  const int doffA = hi * 16;   // A-layout intra-lane K offset
  const int koffB = hi * 16;   // B-layout intra-lane K offset
  const bf16_t* x0 = xb + (size_t)(b * SEQ + row0 + ln) * DIMD;
  const bf16_t* x1 = x0 + (size_t)16 * DIMD;

  // prologue: stage k=0 tile into buffer 0
  cp16_g2l(wsrc + schunk, &bstage[0][srow][schunk]);

  for (int k = 0; k < DIMD; k += 32) {
    const int cur = (k >> 5) & 1;
    wait_async0();            // our async contribution to buf[cur] complete
    __syncthreads();          // everyone's contribution complete; prev reads of buf[cur^1] done
    if (k + 32 < DIMD)
      cp16_g2l(wsrc + (k + 32) + schunk, &bstage[cur ^ 1][srow][schunk]);

    v16bf a0 = frag16(x0 + k + doffA, x0 + k + doffA + 8);
    v16bf a1 = frag16(x1 + k + doffA, x1 + k + doffA + 8);
#pragma unroll
    for (int nt = 0; nt < 4; ++nt) {
      const bf16_t* br = &bstage[cur][nt * 16 + ln][koffB];
      v16bf bb = frag16(br, br + 8);
      acc[0][nt] = wmma_bf16(a0, bb, acc[0][nt]);
      acc[1][nt] = wmma_bf16(a1, bb, acc[1][nt]);
    }
  }

#pragma unroll
  for (int mt = 0; mt < 2; ++mt) {
    const int rowb = row0 + mt * 16 + hi * 8;
#pragma unroll
    for (int nt = 0; nt < 4; ++nt) {
      int e = eb + nt * 16 + ln;
      if (region == 0) {
        if (e < 1024) {                                    // K: [BH,N,64]
          int head = e >> 6, d = e & 63;
          bf16_t* p = kbuf + ((size_t)(b * NHEADS + head) * SEQ) * HD + d;
#pragma unroll
          for (int r = 0; r < 8; ++r) p[(size_t)(rowb + r) * HD] = (bf16_t)acc[mt][nt][r];
        } else {                                           // V -> vT [BH,64,N], packed b128
          int ev = e - 1024; int head = ev >> 6, d = ev & 63;
          v8bf pk;
#pragma unroll
          for (int r = 0; r < 8; ++r) pk[r] = (bf16_t)acc[mt][nt][r];
          *(v8bf*)(vT + ((size_t)(b * NHEADS + head) * HD + d) * SEQ + rowb) = pk;
        }
      } else {                                             // Q (scaled): [BH,N,64]
        int eq = e - 2048; int head = eq >> 6, d = eq & 63;
        bf16_t* p = qbuf + ((size_t)(b * NHEADS + head) * SEQ) * HD + d;
#pragma unroll
        for (int r = 0; r < 8; ++r)
          p[(size_t)(rowb + r) * HD] = (bf16_t)(acc[mt][nt][r] * QSCALE);
      }
    }
  }
}

// ---------------------------------------------------------------- flash attention
// grid (SEQ/128=8, H=16, B=8), block 256. Wave owns 16 query rows.
__global__ __launch_bounds__(256) void attn_kernel(
    const bf16_t* __restrict__ qbuf, const bf16_t* __restrict__ kbuf,
    const bf16_t* __restrict__ vT,   bf16_t* __restrict__ aout /*[B,N,DIM]*/)
{
  __shared__ __align__(16) bf16_t lds[8][16][32];  // per-wave P tile (C->A layout relay)
  const int lane = threadIdx.x & 31;
  const int wave = threadIdx.x >> 5;
  const int ln = lane & 15;
  const int hi = lane >> 4;
  const int h = blockIdx.y, b = blockIdx.z;
  const int bh = b * NHEADS + h;
  const int qrow = blockIdx.x * 128 + wave * 16;

  const bf16_t* qp = qbuf + (size_t)bh * SEQ * HD;
  const bf16_t* kp = kbuf + (size_t)bh * SEQ * HD;
  const bf16_t* vp = vT   + (size_t)bh * HD * SEQ;

  const int doffA = hi * 16;
  const int koffB = hi * 16;

  const bf16_t* qr = qp + (size_t)(qrow + ln) * HD;
  v16bf a0 = frag16(qr + doffA,      qr + doffA + 8);        // head dims 0..31
  v16bf a1 = frag16(qr + 32 + doffA, qr + 32 + doffA + 8);   // head dims 32..63

  float mrow[8], lrow[8];
  v8f acc[4];
#pragma unroll
  for (int r = 0; r < 8; ++r) { mrow[r] = -3.0e38f; lrow[r] = 0.0f; }
#pragma unroll
  for (int dt = 0; dt < 4; ++dt)
#pragma unroll
    for (int r = 0; r < 8; ++r) acc[dt][r] = 0.0f;

  for (int kb = 0; kb < SEQ; kb += 32) {
    const bf16_t* kr0 = kp + (size_t)(kb + ln) * HD;
    const bf16_t* kr1 = kr0 + 16 * HD;
    v8f s0, s1;
#pragma unroll
    for (int r = 0; r < 8; ++r) { s0[r] = 0.0f; s1[r] = 0.0f; }
    {
      v16bf b0 = frag16(kr0 + koffB,      kr0 + koffB + 8);
      v16bf b1 = frag16(kr0 + 32 + koffB, kr0 + 32 + koffB + 8);
      s0 = wmma_bf16(a0, b0, s0);
      s0 = wmma_bf16(a1, b1, s0);
      v16bf b2 = frag16(kr1 + koffB,      kr1 + koffB + 8);
      v16bf b3 = frag16(kr1 + 32 + koffB, kr1 + 32 + koffB + 8);
      s1 = wmma_bf16(a0, b2, s1);
      s1 = wmma_bf16(a1, b3, s1);
    }
    // online softmax; row stats reduced across the 16-lane half holding the row
#pragma unroll
    for (int r = 0; r < 8; ++r) {
      float t = fmaxf(s0[r], s1[r]);
      t = fmaxf(t, __shfl_xor(t, 1, 32));
      t = fmaxf(t, __shfl_xor(t, 2, 32));
      t = fmaxf(t, __shfl_xor(t, 4, 32));
      t = fmaxf(t, __shfl_xor(t, 8, 32));
      float mnew = fmaxf(mrow[r], t);
      float sc = __expf(mrow[r] - mnew);
      mrow[r] = mnew;
      float p0 = __expf(s0[r] - mnew);
      float p1 = __expf(s1[r] - mnew);
      float rs = p0 + p1;
      rs += __shfl_xor(rs, 1, 32);
      rs += __shfl_xor(rs, 2, 32);
      rs += __shfl_xor(rs, 4, 32);
      rs += __shfl_xor(rs, 8, 32);
      lrow[r] = lrow[r] * sc + rs;
#pragma unroll
      for (int dt = 0; dt < 4; ++dt) acc[dt][r] *= sc;
      int row = hi * 8 + r;
      lds[wave][row][ln]      = (bf16_t)p0;
      lds[wave][row][16 + ln] = (bf16_t)p1;
    }
    __asm__ volatile("s_wait_dscnt 0x0" ::: "memory");  // intra-wave LDS RAW
    v16bf pa = frag16(&lds[wave][ln][doffA], &lds[wave][ln][doffA + 8]);
#pragma unroll
    for (int dt = 0; dt < 4; ++dt) {
      const bf16_t* vr = vp + (size_t)(dt * 16 + ln) * SEQ + kb + koffB;
      v16bf bv = frag16(vr, vr + 8);
      acc[dt] = wmma_bf16(pa, bv, acc[dt]);
    }
  }

  float inv[8];
#pragma unroll
  for (int r = 0; r < 8; ++r) inv[r] = 1.0f / lrow[r];
  bf16_t* op = aout + (size_t)b * SEQ * DIMD + (size_t)h * HD;
#pragma unroll
  for (int dt = 0; dt < 4; ++dt)
#pragma unroll
    for (int r = 0; r < 8; ++r) {
      int row = qrow + hi * 8 + r;
      op[(size_t)row * DIMD + dt * 16 + ln] = (bf16_t)(acc[dt][r] * inv[r]);
    }
}

// ---------------------------------------------------------------- output projection
// grid (DIM/64=16, B*SEQ/256=32), block 256. Wave: 32 rows x 64 cols.
// Same LDS-staged, async double-buffered B tiles as the QKV GEMM.
__global__ __launch_bounds__(256) void proj_gemm_kernel(
    const bf16_t* __restrict__ ain,  // [B*N, DIM] bf16
    const bf16_t* __restrict__ pw,   // [DIM, DIM] bf16 (rows = out cols)
    const float*  __restrict__ pb,
    float* __restrict__ out)         // [B*N, DIM] f32
{
  __shared__ __align__(16) bf16_t bstage[2][64][40];

  const int tid  = threadIdx.x;
  const int lane = tid & 31;
  const int wave = tid >> 5;
  const int ln = lane & 15;
  const int hi = lane >> 4;
  const int eb = blockIdx.x * 64;
  const int row0 = blockIdx.y * 256 + wave * 32;
  const int doffA = hi * 16, koffB = hi * 16;

  const int srow   = tid >> 2;
  const int schunk = (tid & 3) * 8;
  const bf16_t* wsrc = pw + (size_t)(eb + srow) * DIMD;

  float bias[4];
#pragma unroll
  for (int nt = 0; nt < 4; ++nt) bias[nt] = pb[eb + nt * 16 + ln];

  v8f acc[2][4];
#pragma unroll
  for (int mt = 0; mt < 2; ++mt)
#pragma unroll
    for (int nt = 0; nt < 4; ++nt)
#pragma unroll
      for (int r = 0; r < 8; ++r) acc[mt][nt][r] = bias[nt];

  const bf16_t* x0 = ain + (size_t)(row0 + ln) * DIMD;
  const bf16_t* x1 = x0 + (size_t)16 * DIMD;

  cp16_g2l(wsrc + schunk, &bstage[0][srow][schunk]);

  for (int k = 0; k < DIMD; k += 32) {
    const int cur = (k >> 5) & 1;
    wait_async0();
    __syncthreads();
    if (k + 32 < DIMD)
      cp16_g2l(wsrc + (k + 32) + schunk, &bstage[cur ^ 1][srow][schunk]);

    v16bf a0 = frag16(x0 + k + doffA, x0 + k + doffA + 8);
    v16bf a1 = frag16(x1 + k + doffA, x1 + k + doffA + 8);
#pragma unroll
    for (int nt = 0; nt < 4; ++nt) {
      const bf16_t* br = &bstage[cur][nt * 16 + ln][koffB];
      v16bf bb = frag16(br, br + 8);
      acc[0][nt] = wmma_bf16(a0, bb, acc[0][nt]);
      acc[1][nt] = wmma_bf16(a1, bb, acc[1][nt]);
    }
  }

#pragma unroll
  for (int mt = 0; mt < 2; ++mt) {
    const int rowb = row0 + mt * 16 + hi * 8;
#pragma unroll
    for (int nt = 0; nt < 4; ++nt) {
      int e = eb + nt * 16 + ln;
#pragma unroll
      for (int r = 0; r < 8; ++r)
        out[(size_t)(rowb + r) * DIMD + e] = acc[mt][nt][r];
    }
  }
}

// ---------------------------------------------------------------- host
extern "C" void kernel_launch(void* const* d_in, const int* in_sizes, int n_in,
                              void* d_out, int out_size, void* d_ws, size_t ws_size,
                              hipStream_t stream) {
  const float* x    = (const float*)d_in[0];
  const int*   cid  = (const int*)  d_in[1];
  const float* kv_w = (const float*)d_in[2];
  const float* kv_b = (const float*)d_in[3];
  const float* sq_w = (const float*)d_in[4];
  const float* sq_b = (const float*)d_in[5];
  const float* cq_w = (const float*)d_in[6];
  const float* cq_b = (const float*)d_in[7];
  const float* pj_w = (const float*)d_in[8];
  const float* pj_b = (const float*)d_in[9];
  float* out = (float*)d_out;

  char* ws = (char*)d_ws;
  size_t off = 0;
  auto carve = [&](size_t elems) -> bf16_t* {
    bf16_t* p = (bf16_t*)(ws + off);
    off = (off + elems * sizeof(bf16_t) + 255) & ~(size_t)255;
    return p;
  };
  const size_t nX  = (size_t)BATCH * SEQ * DIMD;        // 8,388,608
  const size_t nKV = (size_t)2 * DIMD * DIMD;           // 2,097,152
  const size_t nSQ = (size_t)960 * DIMD;                //   983,040
  const size_t nCQ = (size_t)4 * 64 * DIMD;             //   262,144
  const size_t nPW = (size_t)DIMD * DIMD;               // 1,048,576
  const size_t nHB = (size_t)BATCH * NHEADS * SEQ * HD; // 8,388,608

  bf16_t* xb   = carve(nX);
  bf16_t* kvwb = carve(nKV);
  bf16_t* sqwb = carve(nSQ);
  bf16_t* cqwb = carve(nCQ);
  bf16_t* pjwb = carve(nPW);
  bf16_t* qbuf = carve(nHB);
  bf16_t* kbuf = carve(nHB);
  bf16_t* vTb  = carve(nHB);
  bf16_t* aout = carve(nX);

  auto cvt = [&](const float* s, bf16_t* d, size_t n) {
    int ni = (int)n;
    f32_to_bf16_kernel<<<(ni + 255) / 256, 256, 0, stream>>>(s, d, ni);
  };
  cvt(x,    xb,   nX);
  cvt(kv_w, kvwb, nKV);
  cvt(sq_w, sqwb, nSQ);
  cvt(cq_w, cqwb, nCQ);
  cvt(pj_w, pjwb, nPW);

  qkv_gemm_kernel<<<dim3(EQKV / 64, SEQ / 256, BATCH), 256, 0, stream>>>(
      xb, kvwb, kv_b, sqwb, sq_b, cqwb, cq_b, cid, qbuf, kbuf, vTb);

  attn_kernel<<<dim3(SEQ / 128, NHEADS, BATCH), 256, 0, stream>>>(
      qbuf, kbuf, vTb, aout);

  proj_gemm_kernel<<<dim3(DIMD / 64, (BATCH * SEQ) / 256, 1), 256, 0, stream>>>(
      aout, pjwb, pj_b, out);
}
